// HeteroGraph_26809185862282
// MI455X (gfx1250) — compile-verified, matrix operand order checked by reading, not run on hardware
//
#include <hip/hip_runtime.h>
#include <hip/hip_bf16.h>
#include <math.h>

// ---------------------------------------------------------------------------
// HGT forward for gfx1250 (MI455X).
//  - All dense GEMMs on WMMA f16 (f32 accumulate): weights pre-packed into
//    per-lane fragment layout (2x global_load_b128 per B fragment), A tiles
//    staged f32->f16 into LDS (2x ds_load_b128 per A fragment), M-tile = 32
//    rows (2 WMMAs per B fragment).
//  - Scatter-softmax attention in f32 VALU + global atomics (roofline-bound
//    by HBM/atomics, not matrix ops).
// ---------------------------------------------------------------------------

typedef __attribute__((ext_vector_type(16))) _Float16 v16h;
typedef __attribute__((ext_vector_type(8)))  _Float16 v8h;
typedef __attribute__((ext_vector_type(8)))  float    v8f;

#define HIDN   128
#define NHEAD  4
#define DH     32
#define NTYPES 7
#define NREL   14
#define NEDGE  150000
#define NBATCH 1024

// ---------------- fragment helpers (ISA 7.12.2 layouts, wave32) ------------
// Packed-B layout: halfs at ((ct*ksteps + ks)*32 + lane)*16 + j  (j = 0..15)
__device__ __forceinline__ v16h ldg_b_pack(const _Float16* __restrict__ Bp,
                                           int ct, int ks, int ksteps, int lane) {
  const _Float16* p = Bp + (((((long)ct * ksteps + ks) << 5) + lane) << 4);
  const v8h b0 = *(const v8h*)p;
  const v8h b1 = *(const v8h*)(p + 8);
  v16h b;
#pragma unroll
  for (int j = 0; j < 8; ++j) { b[j] = b0[j]; b[j + 8] = b1[j]; }
  return b;
}

// A fragment from an LDS-resident row-major f16 tile (ld = HIDN).
// lane 0-15 / 16-31: rows M=0..15 with K-halves {0..7,16..23} / {8..15,24..31}.
__device__ __forceinline__ v16h lds_a_frag(const _Float16* sA, int k0, int lane) {
  const int r  = lane & 15;
  const int kb = k0 + ((lane >> 4) << 3);
  const v8h a0 = *(const v8h*)(sA + r * HIDN + kb);
  const v8h a1 = *(const v8h*)(sA + r * HIDN + kb + 16);
  v16h a;
#pragma unroll
  for (int j = 0; j < 8; ++j) { a[j] = a0[j]; a[8 + j] = a1[j]; }
  return a;
}

// Stage a 32 x 128 f32 tile into LDS as f16, coalesced (256 threads).
__device__ __forceinline__ void stage_tile32(const float* __restrict__ A, int row0,
                                             int n, _Float16* sA) {
  const int t = threadIdx.x;
  const int r = t >> 4;             // 0..15 -> rows r and r+16
  const int c = (t & 15) << 3;      // 0,8,...,120
#pragma unroll
  for (int h = 0; h < 2; ++h) {
    const int rr = r + h * 16;
    int gr = row0 + rr; if (gr >= n) gr = n - 1;   // tail clamp (stores guarded)
    const float* p = A + (long)gr * HIDN + c;
    v8h hv;
#pragma unroll
    for (int j = 0; j < 8; ++j) hv[j] = (_Float16)p[j];
    *(v8h*)(&sA[rr * HIDN + c]) = hv;
  }
}

// ---------------- weight pre-pack: f32 row-major -> f16 fragment layout -----
// One matrix of K x N per blockIdx.y (matrices contiguous).
__global__ void k_pack_w(const float* __restrict__ W, _Float16* __restrict__ out,
                         int K, int N) {
  const long sz = (long)K * N;
  W   += (long)blockIdx.y * sz;
  out += (long)blockIdx.y * sz;
  const int ksteps = K >> 5;
  const long idx = (long)blockIdx.x * blockDim.x + threadIdx.x;
  if (idx >= sz) return;
  const int  j    = (int)(idx & 15);
  const int  lane = (int)((idx >> 4) & 31);
  const long rest = idx >> 9;
  const int  ks   = (int)(rest % ksteps);
  const int  ct   = (int)(rest / ksteps);
  const int  c    = ct * 16 + (lane & 15);
  const int  k    = ks * 32 + ((lane >> 4) << 3) + (j < 8 ? j : j + 8);
  out[idx] = (_Float16)W[(long)k * N + c];
}

// ---------------- 128-K GEMM: C = A @ W + bias, optional skip blend ---------
// block = 256 threads = 8 waves, 32 rows x 128 cols per block, 8 WMMA/wave.
__global__ void k_gemm128(const float* __restrict__ A, const _Float16* __restrict__ Wp,
                          const float* __restrict__ bias, float* __restrict__ C,
                          const float* __restrict__ Hold, const float* __restrict__ skipg,
                          int n) {
  __shared__ _Float16 sA[32 * HIDN];
  const int lane = threadIdx.x & 31;
  const int wave = threadIdx.x >> 5;
  const int row0 = blockIdx.x << 5;
  if (row0 >= n) return;                 // uniform -> EXEC all-ones at WMMA
  stage_tile32(A, row0, n, sA);
  __syncthreads();
  v8f acc0 = {}, acc1 = {};
#pragma unroll
  for (int ks = 0; ks < 4; ++ks) {
    const v16h b  = ldg_b_pack(Wp, wave, ks, 4, lane);
    const v16h a0 = lds_a_frag(sA, ks * 32, lane);
    const v16h a1 = lds_a_frag(sA + 16 * HIDN, ks * 32, lane);
    acc0 = __builtin_amdgcn_wmma_f32_16x16x32_f16(false, a0, false, b, (short)0, acc0,
                                                  false, false);
    acc1 = __builtin_amdgcn_wmma_f32_16x16x32_f16(false, a1, false, b, (short)0, acc1,
                                                  false, false);
  }
  const int   cn = (wave << 4) + (lane & 15);
  const float bv = bias ? bias[cn] : 0.f;
  float g = 1.f, gi = 0.f;
  if (skipg) { g = 1.f / (1.f + __expf(-skipg[0])); gi = 1.f - g; }
#pragma unroll
  for (int h = 0; h < 2; ++h) {
    const v8f acc = h ? acc1 : acc0;
    const int mb = row0 + h * 16 + ((lane >> 4) << 3);
#pragma unroll
    for (int r = 0; r < 8; ++r) {
      const int m = mb + r;
      if (m < n) {
        float o = acc[r] + bv;
        if (Hold) o = g * o + gi * Hold[(long)m * HIDN + cn];
        C[(long)m * HIDN + cn] = o;
      }
    }
  }
}

// ---------------- per-relation per-head 32x32 transform (K=32) --------------
// Out[n, h*32+e] = sum_d In[n, h*32+d] * Wr[h][d][e]; 32 rows per block.
__global__ void k_rel_transform(const float* __restrict__ In, const _Float16* __restrict__ Wp,
                                float* __restrict__ Out, int n) {
  __shared__ _Float16 sA[32 * HIDN];
  const int lane = threadIdx.x & 31;
  const int wave = threadIdx.x >> 5;    // 8 waves = 4 heads x 2 col-tiles
  const int head = wave >> 1;
  const int ct   = wave & 1;
  const int row0 = blockIdx.x << 5;
  if (row0 >= n) return;
  stage_tile32(In, row0, n, sA);
  __syncthreads();
  const v16h b  = ldg_b_pack(Wp + head * DH * DH, ct, 0, 1, lane);
  const v16h a0 = lds_a_frag(sA, head * DH, lane);
  const v16h a1 = lds_a_frag(sA + 16 * HIDN, head * DH, lane);
  v8f z = {};
  const v8f acc0 = __builtin_amdgcn_wmma_f32_16x16x32_f16(false, a0, false, b, (short)0, z,
                                                          false, false);
  const v8f acc1 = __builtin_amdgcn_wmma_f32_16x16x32_f16(false, a1, false, b, (short)0, z,
                                                          false, false);
  const int cn = head * DH + ct * 16 + (lane & 15);
#pragma unroll
  for (int h = 0; h < 2; ++h) {
    const v8f acc = h ? acc1 : acc0;
    const int mb = row0 + h * 16 + ((lane >> 4) << 3);
#pragma unroll
    for (int r = 0; r < 8; ++r) {
      const int m = mb + r;
      if (m < n) Out[(long)m * HIDN + cn] = acc[r];
    }
  }
}

// ---------------- misc small kernels ---------------------------------------
__global__ void k_fill(float* __restrict__ p, float v, long n) {
  const long i = (long)blockIdx.x * blockDim.x + threadIdx.x;
  if (i < n) p[i] = v;
}

__global__ void k_input_proj(const float* __restrict__ x, const float* __restrict__ W,
                             const float* __restrict__ b, float* __restrict__ h,
                             int n, int f) {
  const int node = blockIdx.x * 2 + (threadIdx.x >> 7);
  const int j = threadIdx.x & 127;
  if (node >= n) return;
  const float* xr = x + (long)node * f;
  float acc = b[j];
  for (int i = 0; i < f; ++i) acc += xr[i] * W[(long)i * HIDN + j];
  h[(long)node * HIDN + j] = acc;
}

__device__ __forceinline__ void atomicMaxF(float* addr, float val) {
  if (val >= 0.f) atomicMax((int*)addr, __float_as_int(val));
  else            atomicMin((unsigned int*)addr, __float_as_uint(val));
}

__global__ void k_edge_alpha(const float* __restrict__ kt, const float* __restrict__ Q,
                             const int* __restrict__ src, const int* __restrict__ dst,
                             const float* __restrict__ prel_r, float* __restrict__ alphaOut,
                             float* __restrict__ amax, int nE, int doff) {
  const int idx = blockIdx.x * blockDim.x + threadIdx.x;
  if (idx >= nE * NHEAD) return;
  const int  e  = idx >> 2, hh = idx & 3;
  const long s  = src[e];
  const long d  = (long)doff + dst[e];
  const float* kp = kt + s * HIDN + hh * DH;
  const float* qp = Q  + d * HIDN + hh * DH;
  float acc = 0.f;
#pragma unroll
  for (int j = 0; j < DH; ++j) acc += kp[j] * qp[j];
  const float a = acc * prel_r[hh] * 0.17677669529663687f;  // 1/sqrt(32)
  alphaOut[(long)e * NHEAD + hh] = a;
  atomicMaxF(&amax[d * NHEAD + hh], a);
}

__global__ void k_edge_msg(const float* __restrict__ vt, const float* __restrict__ alphaIn,
                           const int* __restrict__ src, const int* __restrict__ dst,
                           const float* __restrict__ amax, float* __restrict__ den,
                           float* __restrict__ agg, int nE, int doff) {
  const int idx = blockIdx.x * blockDim.x + threadIdx.x;
  if (idx >= nE * NHEAD) return;
  const int  e  = idx >> 2, hh = idx & 3;
  const long s  = src[e];
  const long d  = (long)doff + dst[e];
  float am = amax[d * NHEAD + hh];
  if (!__builtin_isfinite(am)) am = 0.f;
  const float w = __expf(alphaIn[(long)e * NHEAD + hh] - am);
  atomicAdd(&den[d * NHEAD + hh], w);
  const float* vp = vt + s * HIDN + hh * DH;
  float* ap = agg + d * HIDN + hh * DH;
#pragma unroll
  for (int j = 0; j < DH; ++j) atomicAdd(&ap[j], vp[j] * w);
}

__global__ void k_agg_finish(float* __restrict__ agg, const float* __restrict__ den,
                             long total) {
  const long idx = (long)blockIdx.x * blockDim.x + threadIdx.x;
  if (idx >= total) return;
  const long node = idx >> 7;
  const int  j = (int)(idx & 127);
  float dn = den[node * NHEAD + (j >> 5)];
  dn = dn > 0.f ? dn : 1.f;
  const float x = agg[idx] / dn;
  agg[idx] = 0.5f * x * (1.f + erff(x * 0.70710678118654752f));  // exact GELU
}

__global__ void k_elu_ln(float* __restrict__ Hm, const float* __restrict__ g,
                         const float* __restrict__ bt, long n) {
  const int  lane = threadIdx.x & 31;
  const int  wave = threadIdx.x >> 5;
  const long node = (long)blockIdx.x * 8 + wave;
  if (node >= n) return;
  float* row = Hm + node * HIDN;
  float x[4];
  float s = 0.f;
#pragma unroll
  for (int i = 0; i < 4; ++i) {
    float v = row[lane + i * 32];
    v = v > 0.f ? v : (__expf(v) - 1.f);  // ELU(alpha=1)
    x[i] = v; s += v;
  }
#pragma unroll
  for (int m = 16; m > 0; m >>= 1) s += __shfl_xor(s, m);
  const float mean = s * (1.f / HIDN);
  float vs = 0.f;
#pragma unroll
  for (int i = 0; i < 4; ++i) { const float dd = x[i] - mean; vs += dd * dd; }
#pragma unroll
  for (int m = 16; m > 0; m >>= 1) vs += __shfl_xor(vs, m);
  const float inv = rsqrtf(vs * (1.f / HIDN) + 1e-5f);
#pragma unroll
  for (int i = 0; i < 4; ++i) {
    const int j = lane + i * 32;
    row[j] = (x[i] - mean) * inv * g[j] + bt[j];
  }
}

__global__ void k_pool(const float* __restrict__ Hm, const int* __restrict__ batch,
                       float* __restrict__ pooled, float* __restrict__ cnt, int n) {
  const int  lane = threadIdx.x & 31;
  const int  wave = threadIdx.x >> 5;
  const long node = (long)blockIdx.x * 8 + wave;
  if (node >= n) return;
  const int b = batch[node];
  const float* row = Hm + node * (long)HIDN;
  float* pr = pooled + (long)b * HIDN;
#pragma unroll
  for (int i = 0; i < 4; ++i) atomicAdd(&pr[lane + i * 32], row[lane + i * 32]);
  if (lane == 0) atomicAdd(&cnt[b], 1.f);
}

__global__ void k_heads(const float* __restrict__ pooled, const float* __restrict__ cnt,
                        const float* __restrict__ Wmem, const float* __restrict__ bmem,
                        const float* __restrict__ Wtime, const float* __restrict__ btime,
                        float* __restrict__ out) {
  const int lane = threadIdx.x & 31;
  const int wave = threadIdx.x >> 5;
  const int b = blockIdx.x * 8 + wave;
  if (b >= NBATCH) return;
  float c = cnt[b]; c = c > 0.f ? c : 1.f;
  const float inv = 1.f / c;
  float dm = 0.f, dt = 0.f;
#pragma unroll
  for (int i = 0; i < 4; ++i) {
    const int j = lane + i * 32;
    const float p = pooled[(long)b * HIDN + j] * inv;
    dm += p * Wmem[j];
    dt += p * Wtime[j];
  }
#pragma unroll
  for (int m = 16; m > 0; m >>= 1) { dm += __shfl_xor(dm, m); dt += __shfl_xor(dt, m); }
  if (lane == 0) { out[b] = dm + bmem[0]; out[NBATCH + b] = dt + btime[0]; }
}

// ---------------------------------------------------------------------------
extern "C" void kernel_launch(void* const* d_in, const int* in_sizes, int n_in,
                              void* d_out, int out_size, void* d_ws, size_t ws_size,
                              hipStream_t stream) {
  (void)in_sizes; (void)n_in; (void)out_size; (void)ws_size;
  static const int NNODES[NTYPES] = {200000, 50000, 300000, 150000, 150000, 100000, 100000};
  static const int NFEATS[NTYPES] = {32, 8, 12, 16, 8, 8, 8};
  static const int ESRC[NREL] = {0, 1, 2, 2, 2, 3, 4, 4, 3, 2, 5, 6, 5, 6};
  static const int EDSTR[NREL] = {0, 0, 1, 0, 2, 0, 0, 3, 3, 4, 4, 4, 5, 6};

  long offs[NTYPES + 1]; offs[0] = 0;
  for (int t = 0; t < NTYPES; ++t) offs[t + 1] = offs[t] + NNODES[t];
  const long NTOT = offs[NTYPES];        // 1,050,000

  // ---- inputs (setup_inputs() order) ----
  const float* xin[NTYPES]; const float* Win[NTYPES]; const float* bin[NTYPES];
  for (int t = 0; t < NTYPES; ++t) {
    xin[t] = (const float*)d_in[t * 3 + 0];
    Win[t] = (const float*)d_in[t * 3 + 1];
    bin[t] = (const float*)d_in[t * 3 + 2];
  }
  const float* Wk    = (const float*)d_in[21];
  const float* Wq    = (const float*)d_in[22];
  const float* Wv    = (const float*)d_in[23];
  const float* bk    = (const float*)d_in[24];
  const float* bq    = (const float*)d_in[25];
  const float* bv    = (const float*)d_in[26];
  const float* Watt  = (const float*)d_in[27];
  const float* Wmsg  = (const float*)d_in[28];
  const float* prel  = (const float*)d_in[29];
  const float* Wout  = (const float*)d_in[30];
  const float* bout  = (const float*)d_in[31];
  const float* skip  = (const float*)d_in[32];
  const float* ln_g  = (const float*)d_in[33];
  const float* ln_b  = (const float*)d_in[34];
  const float* Wmem  = (const float*)d_in[35];
  const float* bmem  = (const float*)d_in[36];
  const float* Wtime = (const float*)d_in[37];
  const float* btime = (const float*)d_in[38];
  const int*   EI    = (const int*)d_in[39];   // [14,2,E]
  const int*   batch = (const int*)d_in[40];
  float* out = (float*)d_out;

  // ---- workspace layout ----
  float* ws = (float*)d_ws;
  long p = 0;
  float* h      = ws + p; p += NTOT * HIDN;
  float* Kb     = ws + p; p += NTOT * HIDN;
  float* Qb     = ws + p; p += NTOT * HIDN;
  float* Vb     = ws + p; p += NTOT * HIDN;
  float* agg    = ws + p; p += NTOT * HIDN;
  float* tbuf   = ws + p; p += 300000L * HIDN;                 // kt/vt per relation
  float* alpha  = ws + p; p += (long)NREL * NEDGE * NHEAD;
  float* amax   = ws + p; p += NTOT * NHEAD;
  float* den    = ws + p; p += NTOT * NHEAD;
  float* pooled = ws + p; p += (long)NBATCH * HIDN;
  float* cnt    = ws + p; p += NBATCH;
  // f16 packed weights after the f32 region
  _Float16* hb = (_Float16*)(ws + p);
  long q = 0;
  _Float16* pWk   = hb + q; q += (long)NTYPES * HIDN * HIDN;
  _Float16* pWq   = hb + q; q += (long)NTYPES * HIDN * HIDN;
  _Float16* pWv   = hb + q; q += (long)NTYPES * HIDN * HIDN;
  _Float16* pWout = hb + q; q += (long)NTYPES * HIDN * HIDN;
  _Float16* pWatt = hb + q; q += (long)NREL * NHEAD * DH * DH;
  _Float16* pWmsg = hb + q; q += (long)NREL * NHEAD * DH * DH;

  const int EB = (NEDGE * NHEAD + 255) / 256;
  auto c32 = [](long n) { return (int)((n + 31) / 32); };

  // ---- pre-pack weights into f16 WMMA fragment layout ----
  k_pack_w<<<dim3((HIDN * HIDN + 255) / 256, NTYPES), 256, 0, stream>>>(Wk,   pWk,   HIDN, HIDN);
  k_pack_w<<<dim3((HIDN * HIDN + 255) / 256, NTYPES), 256, 0, stream>>>(Wq,   pWq,   HIDN, HIDN);
  k_pack_w<<<dim3((HIDN * HIDN + 255) / 256, NTYPES), 256, 0, stream>>>(Wv,   pWv,   HIDN, HIDN);
  k_pack_w<<<dim3((HIDN * HIDN + 255) / 256, NTYPES), 256, 0, stream>>>(Wout, pWout, HIDN, HIDN);
  k_pack_w<<<dim3((DH * DH + 255) / 256, NREL * NHEAD), 256, 0, stream>>>(Watt, pWatt, DH, DH);
  k_pack_w<<<dim3((DH * DH + 255) / 256, NREL * NHEAD), 256, 0, stream>>>(Wmsg, pWmsg, DH, DH);

  // ---- input projections ----
  for (int t = 0; t < NTYPES; ++t)
    k_input_proj<<<(NNODES[t] + 1) / 2, 256, 0, stream>>>(
        xin[t], Win[t], bin[t], h + offs[t] * HIDN, NNODES[t], NFEATS[t]);

  // ---- HGT layers ----
  for (int nl = 0; nl < 2; ++nl) {
    for (int t = 0; t < NTYPES; ++t) {
      const long off = offs[t] * HIDN;
      const long wo = (long)t * HIDN * HIDN;
      k_gemm128<<<c32(NNODES[t]), 256, 0, stream>>>(h + off, pWk + wo, bk + t * HIDN,
          Kb + off, nullptr, nullptr, NNODES[t]);
      k_gemm128<<<c32(NNODES[t]), 256, 0, stream>>>(h + off, pWq + wo, bq + t * HIDN,
          Qb + off, nullptr, nullptr, NNODES[t]);
      k_gemm128<<<c32(NNODES[t]), 256, 0, stream>>>(h + off, pWv + wo, bv + t * HIDN,
          Vb + off, nullptr, nullptr, NNODES[t]);
    }
    k_fill<<<(int)((NTOT * NHEAD + 255) / 256), 256, 0, stream>>>(amax, -INFINITY, NTOT * NHEAD);
    k_fill<<<(int)((NTOT * NHEAD + 255) / 256), 256, 0, stream>>>(den, 0.f, NTOT * NHEAD);
    k_fill<<<(int)((NTOT * HIDN + 255) / 256), 256, 0, stream>>>(agg, 0.f, NTOT * HIDN);

    // phase 1: alpha + scatter-max over all relations
    for (int r = 0; r < NREL; ++r) {
      const int s = ESRC[r], d = EDSTR[r];
      k_rel_transform<<<c32(NNODES[s]), 256, 0, stream>>>(
          Kb + offs[s] * HIDN, pWatt + (long)r * NHEAD * DH * DH, tbuf, NNODES[s]);
      k_edge_alpha<<<EB, 256, 0, stream>>>(tbuf, Qb, EI + (long)(r * 2) * NEDGE,
          EI + (long)(r * 2 + 1) * NEDGE, prel + r * NHEAD,
          alpha + (long)r * NEDGE * NHEAD, amax, NEDGE, (int)offs[d]);
    }
    // phase 2: softmax weights + scatter-sum messages
    for (int r = 0; r < NREL; ++r) {
      const int s = ESRC[r], d = EDSTR[r];
      k_rel_transform<<<c32(NNODES[s]), 256, 0, stream>>>(
          Vb + offs[s] * HIDN, pWmsg + (long)r * NHEAD * DH * DH, tbuf, NNODES[s]);
      k_edge_msg<<<EB, 256, 0, stream>>>(tbuf, alpha + (long)r * NEDGE * NHEAD,
          EI + (long)(r * 2) * NEDGE, EI + (long)(r * 2 + 1) * NEDGE,
          amax, den, agg, NEDGE, (int)offs[d]);
    }
    k_agg_finish<<<(int)((NTOT * HIDN + 255) / 256), 256, 0, stream>>>(agg, den, NTOT * HIDN);
    // out-projection + sigmoid(skip) blend, written back into h
    for (int t = 0; t < NTYPES; ++t) {
      const long off = offs[t] * HIDN;
      k_gemm128<<<c32(NNODES[t]), 256, 0, stream>>>(agg + off,
          pWout + (long)t * HIDN * HIDN, bout + t * HIDN, h + off, h + off, skip + t,
          NNODES[t]);
    }
    k_elu_ln<<<(int)((NTOT + 7) / 8), 256, 0, stream>>>(h, ln_g, ln_b, NTOT);
  }

  // ---- mean pool over operator nodes + two linear heads ----
  k_fill<<<(NBATCH * HIDN + NBATCH + 255) / 256, 256, 0, stream>>>(
      pooled, 0.f, (long)NBATCH * HIDN + NBATCH);   // pooled and cnt contiguous
  k_pool<<<(NNODES[0] + 7) / 8, 256, 0, stream>>>(h, batch, pooled, cnt, NNODES[0]);
  k_heads<<<NBATCH / 8, 256, 0, stream>>>(pooled, cnt, Wmem, bmem, Wtime, btime, out);
}
